// CoAtt_58626303590887
// MI455X (gfx1250) — compile-verified, matrix operand order checked by reading, not run on hardware
//
#include <hip/hip_runtime.h>
#include <hip/hip_bf16.h>

// ---------------------------------------------------------------------------
// Problem constants
// ---------------------------------------------------------------------------
#define BB   32
#define LL   256
#define BL   8192          // B*L
#define EW   300
#define EC   100
#define DD   400
#define DP   416           // D padded to multiple of 32
#define HH   256
#define H3   768           // 3*H
#define C2H  512           // 2*H
#define G8H  2048          // 8*H

typedef unsigned short u16;
typedef __attribute__((ext_vector_type(16))) __bf16 v16bf;
typedef __attribute__((ext_vector_type(8)))  float  v8f;

// ---------------------------------------------------------------------------
// Helpers
// ---------------------------------------------------------------------------
__device__ inline u16 f2bf(float f) {
  unsigned int x = __float_as_uint(f);
  unsigned int r = x + 0x7FFFu + ((x >> 16) & 1u);   // round-to-nearest-even
  return (u16)(r >> 16);
}
__device__ inline float bf2f(u16 h) {
  return __uint_as_float(((unsigned int)h) << 16);
}
__device__ inline v8f vzero8() {
  v8f z;
#pragma unroll
  for (int i = 0; i < 8; ++i) z[i] = 0.f;
  return z;
}

// Load a 16x32 bf16 tile (A-operand layout, CDNA5 ISA 7.12.2):
//   lanes 0-15 : row M=lane,    K = {k..k+7, k+16..k+23}
//   lanes 16-31: row M=lane-16, K = {k+8..k+15, k+24..k+31}
// Two 16-byte loads per lane (global_load_b128 / ds_load_b128).
__device__ inline v16bf load_tile(const u16* p0, int ld, int row, int k) {
  int lane = threadIdx.x & 31;
  const u16* p = p0 + (size_t)(row + (lane & 15)) * ld + k + ((lane >> 4) << 3);
  union { v16bf v; uint4 q[2]; } t;
  t.q[0] = *(const uint4*)(p);
  t.q[1] = *(const uint4*)(p + 16);
  return t.v;
}

// Store a 16x16 f32 C tile (C/D layout: lane<16 -> M=0..7, lane>=16 -> M=8..15,
// column = lane&15, one row per VGPR).
__device__ inline void store_tile(float* C, int ldc, int row, int col, v8f acc) {
  int lane = threadIdx.x & 31;
  float* p = C + (size_t)(row + ((lane >> 4) << 3)) * ldc + col + (lane & 15);
#pragma unroll
  for (int v = 0; v < 8; ++v) p[(size_t)v * ldc] = acc[v];
}

__device__ inline v8f wmma_bf16(v16bf a, v16bf b, v8f c) {
  return __builtin_amdgcn_wmma_f32_16x16x32_bf16(false, a, false, b,
                                                 (short)0, c, false, false);
}

// ---------------------------------------------------------------------------
// Tensor Data Mover: stage a 2D tile (32 rows x 768 f32, row stride
// 256*768 elements) from global into LDS. D# per CDNA5 ISA.
// ---------------------------------------------------------------------------
#if __has_builtin(__builtin_amdgcn_tensor_load_to_lds)
#define HAVE_TDM 1
typedef unsigned int tdm_v4u __attribute__((ext_vector_type(4)));
typedef int          tdm_v4i __attribute__((ext_vector_type(4)));
typedef int          tdm_v8i __attribute__((ext_vector_type(8)));

__device__ inline void tdm_load_gi(const float* gsrc, unsigned lds_byte_off) {
  unsigned long long ga = (unsigned long long)(const void*)gsrc;
  tdm_v4u g0;
  g0[0] = 1u;                                    // count=1 (valid), user mode
  g0[1] = lds_byte_off;                          // lds_addr
  g0[2] = (unsigned)(ga & 0xFFFFFFFFu);          // global_addr[31:0]
  g0[3] = (unsigned)((ga >> 32) & 0x01FFFFFFu)   // global_addr[56:32]
          | (2u << 30);                          // type=2 ("image")
  tdm_v8i g1;
  g1[0] = 0x00020000;          // wg_mask=0, data_size=2 (4 bytes)
  g1[1] = (int)(768u << 16);   // tensor_dim0[15:0]=768   (bits 63:48)
  g1[2] = (int)(32u  << 16);   // dim0 hi=0 | tensor_dim1[15:0]=32
  g1[3] = (int)(768u << 16);   // dim1 hi=0 | tile_dim0=768
  g1[4] = 32;                  // tile_dim1=32, tile_dim2=0
  g1[5] = 196608;              // tensor_dim0_stride[31:0] = 256*768
  g1[6] = 0;                   // stride hi | tensor_dim1_stride lo (unused)
  g1[7] = 0;
  tdm_v4i zz = {0, 0, 0, 0};   // groups 2/3 unused (2D tensor)
#if __clang_major__ >= 23
  tdm_v8i z8 = {0, 0, 0, 0, 0, 0, 0, 0};
  __builtin_amdgcn_tensor_load_to_lds(g0, g1, zz, zz, z8, 0);
#else
  __builtin_amdgcn_tensor_load_to_lds(g0, g1, zz, zz, 0);
#endif
}
#endif

// ---------------------------------------------------------------------------
// K1: fp32 -> bf16 weight conversion with K padding (zeros)
// ---------------------------------------------------------------------------
__global__ void k_conv_pad(const float* __restrict__ src, u16* __restrict__ dst,
                           int cols, int colsPad, long long total) {
  long long i = (long long)blockIdx.x * blockDim.x + threadIdx.x;
  if (i >= total) return;
  long long r = i / colsPad;
  int c = (int)(i - r * colsPad);
  dst[i] = (c < cols) ? f2bf(src[r * cols + c]) : (u16)0;
}

// ---------------------------------------------------------------------------
// K2: embedding gather -> bf16 (BL x DP), concat(char, word), zero pad
// ---------------------------------------------------------------------------
__global__ void k_embed(const int* __restrict__ word, const int* __restrict__ chr,
                        const float* __restrict__ wtab, const float* __restrict__ ctab,
                        u16* __restrict__ out) {
  int row = blockIdx.x;                       // 0..BL-1
  int w = word[row], c = chr[row];
  u16* o = out + (size_t)row * DP;
  for (int i = threadIdx.x; i < DP; i += blockDim.x) {
    float v;
    if (i < EC)      v = ctab[(size_t)c * EC + i];
    else if (i < DD) v = wtab[(size_t)w * EW + (i - EC)];
    else             v = 0.f;
    o[i] = f2bf(v);
  }
}

// ---------------------------------------------------------------------------
// K3: generic bf16 WMMA GEMM:  C(M,N) = A(M,K) * Bt(N,K)^T   (fp32 accum)
//   one wave per 16x64 C tile; grid = (N/64, M/16, batch)
//   Ping-pong software pipeline (k-loop unrolled x2): stage-1 loads are in
//   flight while stage-0 WMMAs issue and vice versa; no register rotation.
// ---------------------------------------------------------------------------
__global__ void __launch_bounds__(32)
k_gemm_bf16(const u16* __restrict__ A, int lda, long long sA,
            const u16* __restrict__ Bt, int ldb, long long sB,
            float* __restrict__ C, int ldc, long long sC, int K) {
  A  += (size_t)blockIdx.z * sA;
  Bt += (size_t)blockIdx.z * sB;
  C  += (size_t)blockIdx.z * sC;
  const int row = blockIdx.y * 16;
  const int col = blockIdx.x * 64;
  v8f acc[4];
#pragma unroll
  for (int j = 0; j < 4; ++j) acc[j] = vzero8();

  v16bf a0 = load_tile(A, lda, row, 0);
  v16bf b0[4];
#pragma unroll
  for (int j = 0; j < 4; ++j) b0[j] = load_tile(Bt, ldb, col + j * 16, 0);

  for (int k = 0; k < K; k += 64) {
    const int k1 = k + 32;
    const bool h1 = k1 < K;
    v16bf a1;
    v16bf b1[4];
    if (h1) {                       // stage-1 loads fly over stage-0 WMMAs
      a1 = load_tile(A, lda, row, k1);
#pragma unroll
      for (int j = 0; j < 4; ++j) b1[j] = load_tile(Bt, ldb, col + j * 16, k1);
    }
#pragma unroll
    for (int j = 0; j < 4; ++j) acc[j] = wmma_bf16(a0, b0[j], acc[j]);
    const int k2 = k + 64;
    if (k2 < K) {                   // stage-0 reloads fly over stage-1 WMMAs
      a0 = load_tile(A, lda, row, k2);
#pragma unroll
      for (int j = 0; j < 4; ++j) b0[j] = load_tile(Bt, ldb, col + j * 16, k2);
    }
    if (h1) {
#pragma unroll
      for (int j = 0; j < 4; ++j) acc[j] = wmma_bf16(a1, b1[j], acc[j]);
    }
  }
#pragma unroll
  for (int j = 0; j < 4; ++j) store_tile(C, ldc, row, col + j * 16, acc[j]);
}

// ---------------------------------------------------------------------------
// K4: persistent GRU recurrence. One workgroup (8 waves) per (seq,dir) job.
//   Per step: gh = h @ w_hh^T via WMMA (M=32,N=768,K=256); each wave owns
//   4 (mt,nt) tasks, each task computes the matched (r,z,n) column-tile
//   triple so gates evaluate straight out of the accumulators.
//   Hidden state h lives in LDS (bf16); the next step's gi block
//   (32 x 768 f32) is staged into LDS by the Tensor Data Mover while the
//   that step's matmul runs (TENSORcnt-synchronized).
// ---------------------------------------------------------------------------
struct GruJob {
  const float* gi;     // (BL, 3H) input projection, row = b*L + t
  const u16*   whh;    // (3H, H) bf16
  const float* bih;    // (3H)
  const float* bhh;    // (3H)
  float*       y;      // (B, L, 2H) fp32 or nullptr
  float*       hout;   // final hidden or nullptr
  int          yoff;   // column offset into y (0 fwd / H bwd)
  int          houtld; // row stride of hout
  int          reverse;
  int          pad;
};
struct GruArgs { GruJob job[4]; };

#define GRU_LDS_BYTES (BB * HH * 2 + BB * H3 * 4)   // 16 KB h + 96 KB gi stage

__global__ void __launch_bounds__(256) k_gru(GruArgs args) {
  const GruJob jb = args.job[blockIdx.x];
  extern __shared__ char smem[];
  u16*   h_bf   = (u16*)smem;                       // 32 x 256 bf16
  float* gi_lds = (float*)(smem + BB * HH * 2);     // 32 x 768 f32
  const int tid  = threadIdx.x;
  const int wave = tid >> 5;
  const int lane = tid & 31;
  const int colc = lane & 15;
  const int r0   = (lane >> 4) << 3;

  for (int i = tid; i < BB * HH; i += 256) h_bf[i] = 0;
  __syncthreads();

#ifdef HAVE_TDM
  const unsigned gi_off = (unsigned)(size_t)(void*)gi_lds;  // LDS byte offset
  if (tid < 32) {   // wave 0 drives the TDM (EXEC ignored, one issue per wave)
    const int tt0 = jb.reverse ? (LL - 1) : 0;
    tdm_load_gi(jb.gi + (size_t)tt0 * H3, gi_off);
  }
#endif

  for (int t = 0; t < LL; ++t) {
    const int tt = jb.reverse ? (LL - 1 - t) : t;

#ifndef HAVE_TDM
    // fallback: prefetch next step's gi rows into cache during the matmul
    if (t + 1 < LL && tid < BB) {
      const int tn = jb.reverse ? (LL - 2 - t) : (t + 1);
      __builtin_prefetch(jb.gi + ((size_t)tid * LL + tn) * H3, 0, 1);
    }
#endif

    v8f accR[4], accZ[4], accN[4];
#pragma unroll
    for (int q = 0; q < 4; ++q) {
      const int task = wave * 4 + q;            // 0..31
      const int mt = task >> 4;                 // 2 batch tiles
      const int nt = task & 15;                 // 16 column tiles (j in [0,256))
      v8f ar = vzero8(), az = vzero8(), an = vzero8();
      // ping-pong pipeline over the 3 w_hh tile streams (global, L2-resident)
      v16bf br0 = load_tile(jb.whh, HH, nt * 16, 0);
      v16bf bz0 = load_tile(jb.whh, HH, (16 + nt) * 16, 0);
      v16bf bn0 = load_tile(jb.whh, HH, (32 + nt) * 16, 0);
#pragma unroll
      for (int k = 0; k < HH; k += 64) {
        v16bf br1 = load_tile(jb.whh, HH, nt * 16, k + 32);
        v16bf bz1 = load_tile(jb.whh, HH, (16 + nt) * 16, k + 32);
        v16bf bn1 = load_tile(jb.whh, HH, (32 + nt) * 16, k + 32);
        v16bf a0  = load_tile(h_bf, HH, mt * 16, k);          // LDS
        ar = wmma_bf16(a0, br0, ar);
        az = wmma_bf16(a0, bz0, az);
        an = wmma_bf16(a0, bn0, an);
        if (k + 64 < HH) {
          br0 = load_tile(jb.whh, HH, nt * 16, k + 64);
          bz0 = load_tile(jb.whh, HH, (16 + nt) * 16, k + 64);
          bn0 = load_tile(jb.whh, HH, (32 + nt) * 16, k + 64);
        }
        v16bf a1 = load_tile(h_bf, HH, mt * 16, k + 32);      // LDS
        ar = wmma_bf16(a1, br1, ar);
        az = wmma_bf16(a1, bz1, az);
        an = wmma_bf16(a1, bn1, an);
      }
      accR[q] = ar; accZ[q] = az; accN[q] = an;
    }

#ifdef HAVE_TDM
    if (tid < 32) __builtin_amdgcn_s_wait_tensorcnt(0);  // gi[t] landed in LDS
#endif
    __syncthreads();        // h reads done + gi stage visible to all waves

#pragma unroll
    for (int q = 0; q < 4; ++q) {
      const int task = wave * 4 + q;
      const int mt = task >> 4;
      const int nt = task & 15;
      const int j  = nt * 16 + colc;
      const float bihr = jb.bih[j], bihz = jb.bih[HH + j], bihn = jb.bih[2 * HH + j];
      const float bhhr = jb.bhh[j], bhhz = jb.bhh[HH + j], bhhn = jb.bhh[2 * HH + j];
#pragma unroll
      for (int v = 0; v < 8; ++v) {
        const int b = mt * 16 + r0 + v;
#ifdef HAVE_TDM
        const float* gi = gi_lds + (size_t)b * H3;                 // LDS
#else
        const float* gi = jb.gi + ((size_t)b * LL + tt) * H3;      // global
#endif
        const float ir = gi[j]          + bihr;
        const float iz = gi[HH + j]     + bihz;
        const float in = gi[2 * HH + j] + bihn;
        const float hr = accR[q][v] + bhhr;
        const float hz = accZ[q][v] + bhhz;
        const float hn = accN[q][v] + bhhn;
        const float r = 1.f / (1.f + __expf(-(ir + hr)));
        const float z = 1.f / (1.f + __expf(-(iz + hz)));
        const float n = tanhf(in + r * hn);
        const float hp = bf2f(h_bf[b * HH + j]);
        const float hnew = (1.f - z) * n + z * hp;
        h_bf[b * HH + j] = f2bf(hnew);
        if (jb.y) jb.y[((size_t)b * LL + tt) * C2H + jb.yoff + j] = hnew;
        if (jb.hout && t == LL - 1) jb.hout[(size_t)b * jb.houtld + j] = hnew;
      }
    }
    __syncthreads();        // h updated; all gi[t] reads complete

#ifdef HAVE_TDM
    if (t + 1 < LL && tid < 32) {   // overlap gi[t+1] DMA with next matmul
      const int tn = jb.reverse ? (LL - 2 - t) : (t + 1);
      tdm_load_gi(jb.gi + (size_t)tn * H3, gi_off);
    }
#endif
  }
}

// ---------------------------------------------------------------------------
// K5: fp32 -> bf16 strided row copy (G assembly, ctx slices)
// ---------------------------------------------------------------------------
__global__ void k_copy_bf16(const float* __restrict__ src, long long ss,
                            u16* __restrict__ dst, long long ds, int cols) {
  const float* s = src + (size_t)blockIdx.x * ss;
  u16* d = dst + (size_t)blockIdx.x * ds;
  for (int c = threadIdx.x; c < cols; c += blockDim.x) d[c] = f2bf(s[c]);
}

// K6: Aw3 = (ctx_p * w3) as bf16
__global__ void k_mul_w3(const float* __restrict__ src, const float* __restrict__ w3,
                         u16* __restrict__ dst) {
  const size_t r = blockIdx.x;
  for (int c = threadIdx.x; c < C2H; c += blockDim.x)
    dst[r * C2H + c] = f2bf(src[r * C2H + c] * w3[c]);
}

// K7: per-batch transpose (L x 2H fp32) -> (2H x L bf16)
__global__ void k_transpose_bf16(const float* __restrict__ src, u16* __restrict__ dst) {
  __shared__ float tile[32][33];
  const int b  = blockIdx.z;
  const int c0 = blockIdx.x * 32, r0 = blockIdx.y * 32;
  const int tx = threadIdx.x & 31, ty = threadIdx.x >> 5;
  for (int i = ty; i < 32; i += 8)
    tile[i][tx] = src[((size_t)b * LL + r0 + i) * C2H + c0 + tx];
  __syncthreads();
  for (int i = ty; i < 32; i += 8)
    dst[((size_t)b * C2H + c0 + i) * LL + r0 + tx] = f2bf(tile[tx][i]);
}

// K8: per-batch transpose (L x L fp32) -> (L x L fp32)
__global__ void k_transpose_f32(const float* __restrict__ src, float* __restrict__ dst) {
  __shared__ float tile[32][33];
  const int b  = blockIdx.z;
  const int c0 = blockIdx.x * 32, r0 = blockIdx.y * 32;
  const int tx = threadIdx.x & 31, ty = threadIdx.x >> 5;
  for (int i = ty; i < 32; i += 8)
    tile[i][tx] = src[((size_t)b * LL + r0 + i) * LL + c0 + tx];
  __syncthreads();
  for (int i = ty; i < 32; i += 8)
    dst[((size_t)b * LL + c0 + i) * LL + r0 + tx] = tile[tx][i];
}

// K9: a[row] = dot(x[row,:], w)   (one wave per row, 2H reduction)
__global__ void k_rowdot(const float* __restrict__ x, const float* __restrict__ w,
                         float* __restrict__ out) {
  const int lane = threadIdx.x & 31;
  const int row  = blockIdx.x * 8 + (threadIdx.x >> 5);
  const float* p = x + (size_t)row * C2H;
  float s = 0.f;
  for (int c = lane; c < C2H; c += 32) s += p[c] * w[c];
#pragma unroll
  for (int off = 16; off; off >>= 1) s += __shfl_xor(s, off, 32);
  if (lane == 0) out[row] = s;
}

// K10: row softmax over S (+ rank-1 terms a1[i] + a2[j]), bf16 output
__global__ void k_softmax(const float* __restrict__ S, const float* __restrict__ rowAdd,
                          const float* __restrict__ colAdd, u16* __restrict__ P) {
  const int lane = threadIdx.x & 31;
  const int row  = blockIdx.x * 8 + (threadIdx.x >> 5);   // 0..BL-1
  const int b    = row >> 8;
  const float* s  = S + (size_t)row * LL;
  const float* ca = colAdd + (size_t)b * LL;
  const float ra  = rowAdd[row];
  float v[8];
  float m = -1e30f;
#pragma unroll
  for (int i = 0; i < 8; ++i) {
    v[i] = s[lane + 32 * i] + ra + ca[lane + 32 * i];
    m = fmaxf(m, v[i]);
  }
#pragma unroll
  for (int off = 16; off; off >>= 1) m = fmaxf(m, __shfl_xor(m, off, 32));
  float sum = 0.f;
#pragma unroll
  for (int i = 0; i < 8; ++i) { v[i] = __expf(v[i] - m); sum += v[i]; }
#pragma unroll
  for (int off = 16; off; off >>= 1) sum += __shfl_xor(sum, off, 32);
  const float inv = 1.f / sum;
#pragma unroll
  for (int i = 0; i < 8; ++i) P[(size_t)row * LL + lane + 32 * i] = f2bf(v[i] * inv);
}

// K11: final classifier (32x512)@(512x2)^T -> d_out (fp32, 64 vals)
__global__ void k_final(const float* __restrict__ Pf, const float* __restrict__ W,
                        float* __restrict__ out) {
  const int i = threadIdx.x;
  if (i >= BB * 2) return;
  const int b = i >> 1, c = i & 1;
  float s = 0.f;
  for (int k = 0; k < C2H; ++k) s += Pf[b * C2H + k] * W[c * C2H + k];
  out[i] = s;
}

// ---------------------------------------------------------------------------
// Launcher
// ---------------------------------------------------------------------------
extern "C" void kernel_launch(void* const* d_in, const int* in_sizes, int n_in,
                              void* d_out, int out_size, void* d_ws, size_t ws_size,
                              hipStream_t stream) {
  (void)in_sizes; (void)n_in; (void)out_size; (void)ws_size;
  // ---- inputs --------------------------------------------------------------
  const int*   word_p = (const int*)d_in[0];
  const int*   word_h = (const int*)d_in[1];
  const int*   char_p = (const int*)d_in[2];
  const int*   char_h = (const int*)d_in[3];
  const float* wtab   = (const float*)d_in[4];
  const float* ctab   = (const float*)d_in[5];
  const float* ctx_wih_f = (const float*)d_in[6];
  const float* ctx_whh_f = (const float*)d_in[7];
  const float* ctx_bih_f = (const float*)d_in[8];
  const float* ctx_bhh_f = (const float*)d_in[9];
  const float* ctx_wih_b = (const float*)d_in[10];
  const float* ctx_whh_b = (const float*)d_in[11];
  const float* ctx_bih_b = (const float*)d_in[12];
  const float* ctx_bhh_b = (const float*)d_in[13];
  const float* w_att     = (const float*)d_in[14];
  const float* mod_wih_f = (const float*)d_in[15];
  const float* mod_whh_f = (const float*)d_in[16];
  const float* mod_bih_f = (const float*)d_in[17];
  const float* mod_bhh_f = (const float*)d_in[18];
  const float* mod_wih_b = (const float*)d_in[19];
  const float* mod_whh_b = (const float*)d_in[20];
  const float* mod_bih_b = (const float*)d_in[21];
  const float* mod_bhh_b = (const float*)d_in[22];
  const float* p1_w      = (const float*)d_in[23];

  // ---- workspace layout (bump + phase aliasing; peak ~190 MB) --------------
  size_t off = 0;
  char* base = (char*)d_ws;
  auto alloc = [&](size_t bytes) {
    char* p = base + off;
    off = (off + bytes + 255) & ~(size_t)255;
    return p;
  };
  u16* wihCF = (u16*)alloc((size_t)H3 * DP * 2);
  u16* wihCB = (u16*)alloc((size_t)H3 * DP * 2);
  u16* whhCF = (u16*)alloc((size_t)H3 * HH * 2);
  u16* whhCB = (u16*)alloc((size_t)H3 * HH * 2);
  u16* whhMF = (u16*)alloc((size_t)H3 * HH * 2);
  u16* whhMB = (u16*)alloc((size_t)H3 * HH * 2);
  u16* wihMF = (u16*)alloc((size_t)H3 * G8H * 2);
  u16* wihMB = (u16*)alloc((size_t)H3 * G8H * 2);
  float* ctxP = (float*)alloc((size_t)BL * C2H * 4);
  float* ctxH = (float*)alloc((size_t)BL * C2H * 4);
  float* a1   = (float*)alloc((size_t)BL * 4);
  float* a2   = (float*)alloc((size_t)BL * 4);
  float* Pfin = (float*)alloc((size_t)BB * C2H * 4);
  u16*   G    = (u16*)alloc((size_t)BL * G8H * 2);
  char*  R2   = alloc((size_t)120 << 20);   // reused region, phases below
  (void)R2;

  // phase 1 layout in R2: embeddings + ctx gi
  u16*   embP = (u16*)(R2);
  u16*   embH = (u16*)(R2 + 6815744);
  float* giPF = (float*)(R2 + 13631488);
  float* giPB = (float*)(R2 + 13631488 + 25165824);
  float* giHF = (float*)(R2 + 13631488 + 2 * 25165824);
  float* giHB = (float*)(R2 + 13631488 + 3ll * 25165824);
  // phase 2a layout in R2 (after ctx GRU): attention buffers
  float* S    = (float*)(R2);
  float* St   = (float*)(R2 + 8388608);
  u16*   Pm   = (u16*)(R2 + 16777216);
  u16*   Ptm  = (u16*)(R2 + 20971520);
  u16*   Aw3  = (u16*)(R2 + 25165824);
  u16*   cPT  = (u16*)(R2 + 33554432);
  u16*   cHT  = (u16*)(R2 + 41943040);
  float* p2h  = (float*)(R2 + 50331648);
  float* h2p  = (float*)(R2 + 67108864);
  // phase 2b layout in R2 (after G assembled): mod gi
  float* giMF = (float*)(R2);
  float* giMB = (float*)(R2 + 25165824);

  // ---- 1. weight conversions ----------------------------------------------
  auto conv = [&](const float* s, u16* d, int cols, int colsPad, long long rows) {
    long long tot = rows * colsPad;
    k_conv_pad<<<dim3((unsigned)((tot + 255) / 256)), dim3(256), 0, stream>>>(
        s, d, cols, colsPad, tot);
  };
  conv(ctx_wih_f, wihCF, DD, DP, H3);
  conv(ctx_wih_b, wihCB, DD, DP, H3);
  conv(ctx_whh_f, whhCF, HH, HH, H3);
  conv(ctx_whh_b, whhCB, HH, HH, H3);
  conv(mod_whh_f, whhMF, HH, HH, H3);
  conv(mod_whh_b, whhMB, HH, HH, H3);
  conv(mod_wih_f, wihMF, G8H, G8H, H3);
  conv(mod_wih_b, wihMB, G8H, G8H, H3);

  // ---- 2. embeddings -------------------------------------------------------
  k_embed<<<dim3(BL), dim3(128), 0, stream>>>(word_p, char_p, wtab, ctab, embP);
  k_embed<<<dim3(BL), dim3(128), 0, stream>>>(word_h, char_h, wtab, ctab, embH);

  // ---- 3. ctx GRU input projections (WMMA GEMMs) --------------------------
  dim3 gWave(32);
  k_gemm_bf16<<<dim3(H3 / 64, BL / 16, 1), gWave, 0, stream>>>(
      embP, DP, 0, wihCF, DP, 0, giPF, H3, 0, DP);
  k_gemm_bf16<<<dim3(H3 / 64, BL / 16, 1), gWave, 0, stream>>>(
      embP, DP, 0, wihCB, DP, 0, giPB, H3, 0, DP);
  k_gemm_bf16<<<dim3(H3 / 64, BL / 16, 1), gWave, 0, stream>>>(
      embH, DP, 0, wihCF, DP, 0, giHF, H3, 0, DP);
  k_gemm_bf16<<<dim3(H3 / 64, BL / 16, 1), gWave, 0, stream>>>(
      embH, DP, 0, wihCB, DP, 0, giHB, H3, 0, DP);

  // ---- 4. ctx BiGRU recurrence (4 concurrent persistent workgroups) -------
  GruArgs ga{};
  ga.job[0] = {giPF, whhCF, ctx_bih_f, ctx_bhh_f, ctxP, nullptr, 0,  0, 0, 0};
  ga.job[1] = {giPB, whhCB, ctx_bih_b, ctx_bhh_b, ctxP, nullptr, HH, 0, 1, 0};
  ga.job[2] = {giHF, whhCF, ctx_bih_f, ctx_bhh_f, ctxH, nullptr, 0,  0, 0, 0};
  ga.job[3] = {giHB, whhCB, ctx_bih_b, ctx_bhh_b, ctxH, nullptr, HH, 0, 1, 0};
  k_gru<<<dim3(4), dim3(256), GRU_LDS_BYTES, stream>>>(ga);

  // ---- 5. attention --------------------------------------------------------
  // G[:, 0:512] = ctx_p ; G[:, 512:1024] = ctx_h (also used as einsum B)
  k_copy_bf16<<<dim3(BL), dim3(256), 0, stream>>>(ctxP, C2H, G,       G8H, C2H);
  k_copy_bf16<<<dim3(BL), dim3(256), 0, stream>>>(ctxH, C2H, G + C2H, G8H, C2H);
  k_mul_w3<<<dim3(BL), dim3(256), 0, stream>>>(ctxP, w_att + 2 * C2H, Aw3);
  k_transpose_bf16<<<dim3(C2H / 32, LL / 32, BB), dim3(256), 0, stream>>>(ctxP, cPT);
  k_transpose_bf16<<<dim3(C2H / 32, LL / 32, BB), dim3(256), 0, stream>>>(ctxH, cHT);
  k_rowdot<<<dim3(BL / 8), dim3(256), 0, stream>>>(ctxP, w_att,       a1);
  k_rowdot<<<dim3(BL / 8), dim3(256), 0, stream>>>(ctxH, w_att + C2H, a2);
  // S = (ctx_p*w3) @ ctx_h^T   (batched; rank-1 terms folded into softmax)
  k_gemm_bf16<<<dim3(LL / 64, LL / 16, BB), gWave, 0, stream>>>(
      Aw3, C2H, (long long)LL * C2H,
      G + C2H, G8H, (long long)LL * G8H,
      S, LL, (long long)LL * LL, C2H);
  k_transpose_f32<<<dim3(LL / 32, LL / 32, BB), dim3(256), 0, stream>>>(S, St);
  k_softmax<<<dim3(BL / 8), dim3(256), 0, stream>>>(S,  a1, a2, Pm);
  k_softmax<<<dim3(BL / 8), dim3(256), 0, stream>>>(St, a2, a1, Ptm);
  // p2h = softmax(S) @ ctx_h ;  h2p = softmax(S^T) @ ctx_p
  k_gemm_bf16<<<dim3(C2H / 64, LL / 16, BB), gWave, 0, stream>>>(
      Pm, LL, (long long)LL * LL, cHT, LL, (long long)C2H * LL,
      p2h, C2H, (long long)LL * C2H, LL);
  k_gemm_bf16<<<dim3(C2H / 64, LL / 16, BB), gWave, 0, stream>>>(
      Ptm, LL, (long long)LL * LL, cPT, LL, (long long)C2H * LL,
      h2p, C2H, (long long)LL * C2H, LL);
  k_copy_bf16<<<dim3(BL), dim3(256), 0, stream>>>(p2h, C2H, G + 2 * C2H, G8H, C2H);
  k_copy_bf16<<<dim3(BL), dim3(256), 0, stream>>>(h2p, C2H, G + 3 * C2H, G8H, C2H);

  // ---- 6. mod GRU input projections (largest GEMMs, 8192x768x2048) --------
  k_gemm_bf16<<<dim3(H3 / 64, BL / 16, 1), gWave, 0, stream>>>(
      G, G8H, 0, wihMF, G8H, 0, giMF, H3, 0, G8H);
  k_gemm_bf16<<<dim3(H3 / 64, BL / 16, 1), gWave, 0, stream>>>(
      G, G8H, 0, wihMB, G8H, 0, giMB, H3, 0, G8H);

  // ---- 7. mod BiGRU recurrence (final hidden states only) -----------------
  GruArgs gm{};
  gm.job[0] = {giMF, whhMF, mod_bih_f, mod_bhh_f, nullptr, Pfin,      0, C2H, 0, 0};
  gm.job[1] = {giMB, whhMB, mod_bih_b, mod_bhh_b, nullptr, Pfin + HH, 0, C2H, 1, 0};
  gm.job[2] = gm.job[0];  // unused (grid is 2)
  gm.job[3] = gm.job[1];
  k_gru<<<dim3(2), dim3(256), GRU_LDS_BYTES, stream>>>(gm);

  // ---- 8. classifier -------------------------------------------------------
  k_final<<<dim3(1), dim3(64), 0, stream>>>(Pfin, p1_w, (float*)d_out);
}
